// LightHANet_82197084111491
// MI455X (gfx1250) — matrix operand order repaired and checked by reading, not compiled, not run on hardware
//
#include <hip/hip_runtime.h>
#include <hip/hip_bf16.h>
#include <stdint.h>

// ---------------- problem constants ----------------
#define Bn     4
#define Cn     256
#define Hn     96
#define Wn     96
#define HWn    (Hn * Wn)          // 9216
#define NPIX   (Bn * HWn)         // 36864
#define KK     5
#define TAPS   (KK * KK)          // 25
#define OFFC   (2 * TAPS)         // 50
#define MIDn   32                 // C / RED
#define NGRP   8
#define CPG    (Cn / NGRP)        // 32
#define WPAD   98                 // H+2 / W+2 padded
#define WP2    (WPAD * WPAD)      // 9604
#define KTOT   (Cn * 9)           // 2304 implicit-GEMM K
#define KT     32                 // K per wmma step
#define COT    128                // co tile per block
#define PXT    128                // pixel tile per block
#define PXTILES (NPIX / PXT)      // 288

// ---------------- workspace layout (bytes) ----------------
#define WS_OFFSET   ((size_t)0)                                  // 4*50*9216 f32 = 7,372,800
#define WS_ALIGNED  ((size_t)7372800)                            // 4*256*9216 f32 = 37,748,736
#define WS_POOLED   ((size_t)45121536)                           // 1024 f32
#define WS_GATE     ((size_t)45125632)                           // 1024 f32
#define WS_STATS    ((size_t)45129728)                           // 64 f32
#define WS_WBF      ((size_t)45130240)                           // 2304*256 u16 = 1,179,648
#define WS_FPAD     ((size_t)46309888)                           // 4*256*98*98 u16 = 19,668,992
#define WS_FPAD_BYTES ((size_t)Bn * Cn * WPAD * WPAD * 2)

typedef __attribute__((ext_vector_type(16))) __bf16 v16bf;
typedef __attribute__((ext_vector_type(8)))  float  v8f;
typedef __attribute__((ext_vector_type(4)))  unsigned int u32x4;
typedef __attribute__((ext_vector_type(8)))  int          i32x8;
typedef __attribute__((ext_vector_type(4)))  int          i32x4;

#if defined(__has_builtin)
#  if __has_builtin(__builtin_amdgcn_tensor_load_to_lds)
#    define HAS_TDM 1
#  else
#    define HAS_TDM 0
#  endif
#else
#  define HAS_TDM 0
#endif

struct alignas(16) Half8 { unsigned int u[4]; };
union FragCvt { struct { Half8 lo, hi; } h; v16bf v; };

__device__ __forceinline__ unsigned short f2bf(float f) {
    unsigned int u = __builtin_bit_cast(unsigned int, f);
    unsigned int r = u + 0x7FFFu + ((u >> 16) & 1u);   // round-to-nearest-even
    return (unsigned short)(r >> 16);
}

// ============ 1) offset = 1x1 conv(concat(cur,hist)) + bias ============
__global__ __launch_bounds__(256) void offset_conv(
    const float* __restrict__ cur, const float* __restrict__ hist,
    const float* __restrict__ w,   const float* __restrict__ bias,
    float* __restrict__ offset)
{
    int gpix = blockIdx.x * 256 + threadIdx.x;          // < NPIX
    int b = gpix / HWn, hw = gpix % HWn;
    float acc[OFFC];
    #pragma unroll
    for (int j = 0; j < OFFC; ++j) acc[j] = bias[j];    // uniform -> s_load
    const float* cb = cur  + (size_t)b * Cn * HWn + hw;
    const float* hb = hist + (size_t)b * Cn * HWn + hw;
    for (int c = 0; c < Cn; ++c) {
        float xv = cb[c * HWn];
        #pragma unroll
        for (int j = 0; j < OFFC; ++j) acc[j] += xv * w[j * (2 * Cn) + c];
    }
    for (int c = 0; c < Cn; ++c) {
        float xv = hb[c * HWn];
        #pragma unroll
        for (int j = 0; j < OFFC; ++j) acc[j] += xv * w[j * (2 * Cn) + Cn + c];
    }
    float* ob = offset + (size_t)b * OFFC * HWn + hw;
    #pragma unroll
    for (int j = 0; j < OFFC; ++j) ob[j * HWn] = acc[j];
}

// ============ 2) depthwise deformable conv ============
#define DPIX 64
__global__ __launch_bounds__(DPIX) void deform_kernel(
    const float* __restrict__ hist, const float* __restrict__ offset,
    const float* __restrict__ dw,   float* __restrict__ aligned)
{
    __shared__ int   sIdx[DPIX * TAPS * 4];   // 25.6 KB
    __shared__ float sW  [DPIX * TAPS * 4];   // 25.6 KB
    int tid  = threadIdx.x;
    int gpix = blockIdx.x * DPIX + tid;       // blocks of 64 stay within one b
    int b = gpix / HWn, hw = gpix % HWn;
    int h = hw / Wn,    wq = hw % Wn;

    const float* offp = offset + (size_t)b * OFFC * HWn + hw;
    #pragma unroll
    for (int k = 0; k < TAPS; ++k) {
        float dy = offp[(2 * k)     * HWn];
        float dx = offp[(2 * k + 1) * HWn];
        float py = (float)(h + k / KK - 2) + dy;
        float px = (float)(wq + k % KK - 2) + dx;
        float fy = floorf(py), fx = floorf(px);
        int y0 = (int)fy, x0 = (int)fx;
        float wy1 = py - fy, wx1 = px - fx;
        float wy0 = 1.f - wy1, wx0 = 1.f - wx1;
        int base = (tid * TAPS + k) * 4;
        int ys[2] = { y0, y0 + 1 };
        int xs[2] = { x0, x0 + 1 };
        float wy[2] = { wy0, wy1 };
        float wx[2] = { wx0, wx1 };
        #pragma unroll
        for (int cy = 0; cy < 2; ++cy)
        #pragma unroll
        for (int cx = 0; cx < 2; ++cx) {
            int yi = ys[cy], xi = xs[cx];
            bool ok = (yi >= 0) & (yi < Hn) & (xi >= 0) & (xi < Wn);
            sIdx[base + cy * 2 + cx] = ok ? (yi * Wn + xi) : 0;
            sW  [base + cy * 2 + cx] = ok ? (wy[cy] * wx[cx]) : 0.f;
        }
    }
    const float* hbase = hist    + (size_t)b * Cn * HWn;
    float*       abase = aligned + (size_t)b * Cn * HWn + hw;
    for (int c = 0; c < Cn; ++c) {
        const float* hc  = hbase + c * HWn;
        const float* dwc = dw + c * TAPS;                // uniform -> s_load
        float acc = 0.f;
        #pragma unroll 5
        for (int k = 0; k < TAPS; ++k) {
            int bi = (tid * TAPS + k) * 4;
            float s = sW[bi + 0] * hc[sIdx[bi + 0]]
                    + sW[bi + 1] * hc[sIdx[bi + 1]]
                    + sW[bi + 2] * hc[sIdx[bi + 2]]
                    + sW[bi + 3] * hc[sIdx[bi + 3]];
            acc += dwc[k] * s;
        }
        abase[c * HWn] = acc;
    }
}

// ============ 3) pooled[b,c] = mean(cur - aligned) ============
__global__ __launch_bounds__(256) void pool_reduce(
    const float* __restrict__ cur, const float* __restrict__ aligned,
    float* __restrict__ pooled)
{
    int bc = blockIdx.x;
    const float* c0 = cur     + (size_t)bc * HWn;
    const float* a0 = aligned + (size_t)bc * HWn;
    float s = 0.f;
    for (int i = threadIdx.x; i < HWn; i += 256) s += c0[i] - a0[i];
    __shared__ float red[256];
    red[threadIdx.x] = s; __syncthreads();
    for (int st = 128; st > 0; st >>= 1) {
        if (threadIdx.x < st) red[threadIdx.x] += red[threadIdx.x + st];
        __syncthreads();
    }
    if (threadIdx.x == 0) pooled[bc] = red[0] * (1.f / (float)HWn);
}

// ============ 4) SE gate ============
__global__ __launch_bounds__(256) void se_kernel(
    const float* __restrict__ pooled, const float* __restrict__ w1,
    const float* __restrict__ w2, float* __restrict__ gate)
{
    __shared__ float sp[Bn * Cn];
    __shared__ float sa[Bn * MIDn];
    for (int i = threadIdx.x; i < Bn * Cn; i += 256) sp[i] = pooled[i];
    __syncthreads();
    if (threadIdx.x < Bn * MIDn) {
        int b = threadIdx.x / MIDn, m = threadIdx.x % MIDn;
        float s = 0.f;
        for (int c = 0; c < Cn; ++c) s += sp[b * Cn + c] * w1[m * Cn + c];
        sa[threadIdx.x] = fmaxf(s, 0.f);
    }
    __syncthreads();
    for (int idx = threadIdx.x; idx < Bn * Cn; idx += 256) {
        int b = idx / Cn, c = idx % Cn;
        float s = 0.f;
        #pragma unroll
        for (int m = 0; m < MIDn; ++m) s += sa[b * MIDn + m] * w2[c * MIDn + m];
        gate[idx] = 1.f / (1.f + __expf(-s));
    }
}

// ============ 5) fused = cur + aligned*g  ->  bf16, zero-padded 98x98 ============
__global__ __launch_bounds__(256) void fuse_pad(
    const float* __restrict__ cur, const float* __restrict__ aligned,
    const float* __restrict__ gate, unsigned short* __restrict__ fpad)
{
    size_t i = (size_t)blockIdx.x * 256 + threadIdx.x;   // < B*C*H*W
    int bc = (int)(i / HWn), hw = (int)(i % HWn);
    int h = hw / Wn, wq = hw % Wn;
    float v = cur[i] + aligned[i] * gate[bc];
    fpad[((size_t)bc * WPAD + h + 1) * WPAD + (wq + 1)] = f2bf(v);
}

// ============ 6) repack out_w -> bf16 [co][k], k = (ky*3+kx)*256 + ci ============
__global__ __launch_bounds__(256) void wconv_kernel(
    const float* __restrict__ out_w, unsigned short* __restrict__ wbf)
{
    int i = blockIdx.x * 256 + threadIdx.x;              // < 256*2304
    int co = i / KTOT, k = i % KTOT;
    int s = k >> 8, ci = k & 255;
    wbf[i] = f2bf(out_w[((size_t)co * Cn + ci) * 9 + s]);
}

// ============ 7) 3x3 conv as implicit GEMM, bf16 WMMA + TDM A-tile ============
__global__ __launch_bounds__(256) void conv3x3_wmma(
    const unsigned short* __restrict__ fpad,
    const unsigned short* __restrict__ wbf,
    float* __restrict__ y)
{
    __shared__ __align__(16) unsigned short lA[COT * KT];   // [co_l][k]  8 KB
    __shared__ __align__(16) unsigned short lB[PXT * KT];   // [px_l][k]  8 KB

    int tid  = threadIdx.x;
    int lane = tid & 31;
    int wave = tid >> 5;
    int tile = blockIdx.x;                 // 0..287  pixel tile
    int b    = tile / (HWn / PXT);
    int pix0 = (tile % (HWn / PXT)) * PXT;
    int co0  = blockIdx.y * COT;

    int cp = wave & 3;                     // co pair 0..3 -> co chunks {2cp, 2cp+1}
    int pg = wave >> 2;                    // pixel group 0..1 -> px chunks pg*4..+3

    v8f acc[2][4] = {};

    // ---- B-tile gather mapping + hoisted per-pixel offsets ----
    int b_kk = tid >> 3;                   // 0..31  (k within chunk)
    int b_ps = tid & 7;
    int hwoff[16];
    #pragma unroll
    for (int i = 0; i < 16; ++i) {
        int px = pix0 + b_ps + (i << 3);
        int h = px / Wn, wq = px % Wn;
        hwoff[i] = (h + 1) * WPAD + (wq + 1);   // center tap offset in padded image
    }
    const unsigned short* fb = fpad + (size_t)b * Cn * WP2;

#if HAS_TDM && defined(__AMDGCN__)
    // ---- TDM descriptor: 2D tile (tile_dim0=32 k, tile_dim1=128 co) of the
    //      bf16 weight tensor [256 co][2304 k], row stride 2304 elements. ----
    unsigned int ldsa = (unsigned int)(uintptr_t)&lA[0];    // LDS byte offset
    i32x8 g1 = { 0x00010000,                 // workgroup_mask=0, data_size=1 (2B)
                 (int)((KTOT & 0xFFFF) << 16),          // tensor_dim0[15:0]
                 (int)((Cn & 0xFFFF) << 16),            // dim0[31:16]=0 | tensor_dim1[15:0]
                 (int)(KT << 16),                       // dim1[31:16]=0 | tile_dim0
                 (int)COT,                              // tile_dim1 | tile_dim2=0
                 (int)KTOT,                             // tensor_dim0_stride[31:0]
                 0, 0 };
    i32x4 gz = { 0, 0, 0, 0 };
#if defined(__clang_major__) && (__clang_major__ >= 23)
    i32x8 gz8 = { 0, 0, 0, 0, 0, 0, 0, 0 };
#endif
#else
    // fallback A-copy mapping: 16 contiguous halves per thread
    int a_col = tid >> 1;                  // co_l 0..127
    int a_kk  = (tid & 1) << 4;            // 0 or 16
#endif

    for (int k0 = 0; k0 < KTOT; k0 += KT) {
        int s  = k0 >> 8;                  // tap index, constant over the chunk
        int ky = s / 3, kx = s % 3;
        int ci = (k0 & 255) + b_kk;
        int dlt = (ky - 1) * WPAD + (kx - 1);

        __syncthreads();                   // previous tiles fully consumed

#if HAS_TDM && defined(__AMDGCN__)
        if (wave == 0) {
            unsigned long long ga =
                (unsigned long long)(uintptr_t)(wbf + (size_t)co0 * KTOT + k0);
            u32x4 g0 = { 1u,                                   // count=1, user mode
                         ldsa,                                 // lds_addr
                         (unsigned int)ga,                     // global_addr[31:0]
                         (unsigned int)((ga >> 32) & 0x1FFFFFFu) | 0x80000000u }; // type=2
#if defined(__clang_major__) && (__clang_major__ >= 23)
            __builtin_amdgcn_tensor_load_to_lds(g0, g1, gz, gz, gz8, 0);
#else
            __builtin_amdgcn_tensor_load_to_lds(g0, g1, gz, gz, 0);
#endif
        }
#else
        {
            const Half8* src = (const Half8*)&wbf[(size_t)(co0 + a_col) * KTOT + k0 + a_kk];
            Half8* dst = (Half8*)&lA[a_col * KT + a_kk];
            dst[0] = src[0];
            dst[1] = src[1];
        }
#endif
        // ---- stage B: activations [128 px][32 k] (address math hoisted) ----
        {
            const unsigned short* fc = fb + (size_t)ci * WP2;
            #pragma unroll
            for (int i = 0; i < 16; ++i)
                lB[(b_ps + (i << 3)) * KT + b_kk] = fc[hwoff[i] + dlt];
        }
#if HAS_TDM && defined(__AMDGCN__)
        if (wave == 0) __builtin_amdgcn_s_wait_tensorcnt(0);
#endif
        __syncthreads();

        // ---- A fragments (16-bit A layout: half-wave k 0-7/16-23 vs 8-15/24-31) ----
        FragCvt fa[2];
        {
            int m  = lane & 15;
            int kb = (lane >> 4) << 3;
            #pragma unroll
            for (int a = 0; a < 2; ++a) {
                int row = (cp * 2 + a) * 16 + m;
                fa[a].h.lo = *(const Half8*)&lA[row * KT + kb];
                fa[a].h.hi = *(const Half8*)&lA[row * KT + 16 + kb];
            }
        }
        // ---- B fragments (half-wave k 0-7/16-23 vs 8-15/24-31), reused across both co chunks ----
        {
            int n   = lane & 15;
            int kb2 = (lane >> 4) << 3;
            #pragma unroll
            for (int j = 0; j < 4; ++j) {
                FragCvt fbv;
                const Half8* pb = (const Half8*)&lB[((pg * 4 + j) * 16 + n) * KT + kb2];
                fbv.h.lo = pb[0];
                fbv.h.hi = pb[2];
                #pragma unroll
                for (int a = 0; a < 2; ++a)
                    acc[a][j] = __builtin_amdgcn_wmma_f32_16x16x32_bf16(
                        false, fa[a].v, false, fbv.v, (short)0, acc[a][j], false, false);
            }
        }
    }

    // ---- epilogue: C layout (lanes 0-15: M=r, lanes 16-31: M=r+8; N=lane&15) ----
    int n    = lane & 15;
    int mofs = (lane >> 4) << 3;
    #pragma unroll
    for (int a = 0; a < 2; ++a) {
        #pragma unroll
        for (int j = 0; j < 4; ++j) {
            int px = pix0 + (pg * 4 + j) * 16 + n;
            #pragma unroll
            for (int r = 0; r < 8; ++r) {
                int co = co0 + (cp * 2 + a) * 16 + r + mofs;
                y[((size_t)b * Cn + co) * HWn + px] = acc[a][j][r];
            }
        }
    }
}

// ============ 8) GroupNorm stats ============
__global__ __launch_bounds__(256) void gn_stats(
    const float* __restrict__ y, float* __restrict__ stats)
{
    int blk = blockIdx.x;                 // b*8 + group
    int b = blk >> 3, gr = blk & 7;
    const float* base = y + ((size_t)b * Cn + gr * CPG) * HWn;
    const int N = CPG * HWn;
    float s = 0.f, q = 0.f;
    for (int i = threadIdx.x; i < N; i += 256) {
        float v = base[i]; s += v; q += v * v;
    }
    __shared__ float rs[256], rq[256];
    rs[threadIdx.x] = s; rq[threadIdx.x] = q; __syncthreads();
    for (int st = 128; st > 0; st >>= 1) {
        if (threadIdx.x < st) { rs[threadIdx.x] += rs[threadIdx.x + st];
                                rq[threadIdx.x] += rq[threadIdx.x + st]; }
        __syncthreads();
    }
    if (threadIdx.x == 0) {
        float mean = rs[0] / (float)N;
        float var  = rq[0] / (float)N - mean * mean;
        stats[blk * 2]     = mean;
        stats[blk * 2 + 1] = rsqrtf(var + 1e-5f);
    }
}

// ============ 9) GN apply + ReLU (in place on d_out) ============
__global__ __launch_bounds__(256) void gn_apply(
    float* __restrict__ y, const float* __restrict__ stats,
    const float* __restrict__ gamma, const float* __restrict__ beta)
{
    size_t i = (size_t)blockIdx.x * 256 + threadIdx.x;
    int bc = (int)(i / HWn);
    int b = bc >> 8, c = bc & 255;
    int st = (b * NGRP + (c >> 5)) * 2;
    float v = (y[i] - stats[st]) * stats[st + 1] * gamma[c] + beta[c];
    y[i] = fmaxf(v, 0.f);
}

// ---------------- host launcher ----------------
extern "C" void kernel_launch(void* const* d_in, const int* in_sizes, int n_in,
                              void* d_out, int out_size, void* d_ws, size_t ws_size,
                              hipStream_t stream) {
    const float* cur      = (const float*)d_in[0];
    const float* hist     = (const float*)d_in[1];
    const float* offset_w = (const float*)d_in[2];
    const float* offset_b = (const float*)d_in[3];
    const float* deform_w = (const float*)d_in[4];
    const float* se_w1    = (const float*)d_in[5];
    const float* se_w2    = (const float*)d_in[6];
    const float* out_w    = (const float*)d_in[7];
    const float* gn_gamma = (const float*)d_in[8];
    const float* gn_beta  = (const float*)d_in[9];

    char* ws = (char*)d_ws;
    float*          offset  = (float*)(ws + WS_OFFSET);
    float*          aligned = (float*)(ws + WS_ALIGNED);
    float*          pooled  = (float*)(ws + WS_POOLED);
    float*          gate    = (float*)(ws + WS_GATE);
    float*          stats   = (float*)(ws + WS_STATS);
    unsigned short* wbf     = (unsigned short*)(ws + WS_WBF);
    unsigned short* fpad    = (unsigned short*)(ws + WS_FPAD);
    float*          y       = (float*)d_out;

    offset_conv<<<NPIX / 256, 256, 0, stream>>>(cur, hist, offset_w, offset_b, offset);
    deform_kernel<<<NPIX / DPIX, DPIX, 0, stream>>>(hist, offset, deform_w, aligned);
    pool_reduce<<<Bn * Cn, 256, 0, stream>>>(cur, aligned, pooled);
    se_kernel<<<1, 256, 0, stream>>>(pooled, se_w1, se_w2, gate);
    hipMemsetAsync(fpad, 0, WS_FPAD_BYTES, stream);
    fuse_pad<<<(Bn * Cn * HWn) / 256, 256, 0, stream>>>(cur, aligned, gate, fpad);
    wconv_kernel<<<(Cn * KTOT) / 256, 256, 0, stream>>>(out_w, wbf);
    conv3x3_wmma<<<dim3(PXTILES, Cn / COT), 256, 0, stream>>>(fpad, wbf, y);
    gn_stats<<<Bn * NGRP, 256, 0, stream>>>(y, stats);
    gn_apply<<<(Bn * Cn * HWn) / 256, 256, 0, stream>>>(y, stats, gn_gamma, gn_beta);
}
